// GCNLayer_33114197852234
// MI455X (gfx1250) — matrix operand order repaired and checked by reading, not compiled, not run on hardware
//
#include <hip/hip_runtime.h>
#include <hip/hip_bf16.h>

// CDNA5 / gfx1250 WMMA operand types
typedef __attribute__((ext_vector_type(16))) __bf16 v16bf;
typedef __attribute__((ext_vector_type(8)))  float  v8f;
typedef __attribute__((ext_vector_type(4)))  int    v4i;   // b128 payload type

// Async global->LDS path (GLOBAL_LOAD_ASYNC_TO_LDS_B128, ASYNCcnt)
#if defined(__has_builtin)
#  if __has_builtin(__builtin_amdgcn_global_load_async_to_lds_b128) && \
      __has_builtin(__builtin_amdgcn_s_wait_asynccnt)
#    define USE_ASYNC_LDS 1
#  endif
#endif
#ifndef USE_ASYNC_LDS
#  define USE_ASYNC_LDS 0
#endif

#define N_NODES 8192
#define FEAT    512

// Block tile 128(M) x 128(N), K-step 32. 256 threads = 8 wave32s in a 4x2
// grid; each wave owns a 32x64 patch = 2x4 v_wmma_f32_16x16x32_bf16 accums.
#define BM 128
#define BN 128
#define BK 32
#define LDA 40   // padded LDS row stride (bf16 elems): 80B, 16B-aligned rows
#define LDB 40

// ---------------------------------------------------------------------------
// Kernel 1: d[i] = rsqrt(1 + sum_j adj[i,j])   (row degree of adj + I)
// ---------------------------------------------------------------------------
__global__ __launch_bounds__(256)
void gcn_degree_kernel(const float* __restrict__ adj, float* __restrict__ dinv) {
    const int row = blockIdx.x;
    const float4* r4 = reinterpret_cast<const float4*>(adj + (size_t)row * N_NODES);
    float s = 0.f;
    for (int j = threadIdx.x; j < N_NODES / 4; j += 256) {
        float4 v = r4[j];
        s += (v.x + v.y) + (v.z + v.w);
    }
    __shared__ float red[256];
    red[threadIdx.x] = s;
    __syncthreads();
    for (int off = 128; off > 0; off >>= 1) {
        if (threadIdx.x < off) red[threadIdx.x] += red[threadIdx.x + off];
        __syncthreads();
    }
    if (threadIdx.x == 0) {
        float rs = red[0] + 1.0f;                 // + identity
        dinv[row] = (rs > 0.f) ? rsqrtf(rs) : 0.f;
    }
}

// ---------------------------------------------------------------------------
// Fragment loaders (wave32; A/B in this code's K-contiguous order, not the ISA table order)
//   A 16x32 bf16: lane -> M = lane&15; K = 16*(lane>>4) + i  (16 K-contiguous)
//   B 32x16 bf16: lane -> N = lane&15; K = 16*(lane>>4) + i  (16 K-contiguous)
//   C/D 16x16 f32: lane -> N = lane&15; VGPR r -> M = 8*(lane>>4) + r.
// ---------------------------------------------------------------------------
__device__ __forceinline__ v16bf load_a_frag(const __bf16* __restrict__ ap, int lane) {
    const int kk = (lane >> 4) << 4;     // 0 or 16
    v16bf a;
#pragma unroll
    for (int i = 0; i < 16; ++i) a[i] = ap[kk + i];
    return a;
}

__device__ __forceinline__ v16bf load_b_frag(const __bf16* __restrict__ bp) {
    v16bf b;                              // 16 K-contiguous bf16 = 2x ds_load_b128
#pragma unroll
    for (int i = 0; i < 16; ++i) b[i] = bp[i];
    return b;
}

// ---------------------------------------------------------------------------
// Kernel 2: supportT[f, j] = bf16( d[j] * (x @ W)[j, f] )   (stored transposed)
// ---------------------------------------------------------------------------
__global__ __launch_bounds__(256)
void gcn_xw_kernel(const float* __restrict__ x,
                   const float* __restrict__ w,
                   const float* __restrict__ dinv,
                   __bf16* __restrict__ supportT) {
    __shared__ __bf16 As[BM * LDA];   // x tile   [m][k]
    __shared__ __bf16 Bs[BN * LDB];   // W tile   [n][k] (K-contiguous per column)

    const int tid  = threadIdx.x;
    const int wave = tid >> 5, lane = tid & 31;
    const int wr = wave & 3, wc = wave >> 2;   // 4x2 wave grid
    const int cb = blockIdx.x * BN;            // x = column tile (L2 band reuse)
    const int rb = blockIdx.y * BM;

    float4 pa[4], pw[4];                       // register double-buffer
    auto load_tile = [&](int kb) {
#pragma unroll
        for (int t = 0; t < 4; ++t) {
            int idx = tid + t * 256;
            int r = idx >> 3, c4 = (idx & 7) << 2;
            pa[t] = *reinterpret_cast<const float4*>(
                x + (size_t)(rb + r) * FEAT + kb + c4);
        }
        int k = tid & 31, ng = tid >> 5;       // ng -> 16 columns each
        const float4* srcw = reinterpret_cast<const float4*>(
            w + (size_t)(kb + k) * FEAT + cb + ng * 16);
#pragma unroll
        for (int t = 0; t < 4; ++t) pw[t] = srcw[t];
    };

    v8f acc[2][4] = {};
    load_tile(0);

    for (int kb = 0; kb < FEAT; kb += BK) {
        __syncthreads();
        // ---- commit staged regs to LDS (fp32 -> bf16) ----
#pragma unroll
        for (int t = 0; t < 4; ++t) {
            int idx = tid + t * 256;
            int r = idx >> 3, c4 = (idx & 7) << 2;
            As[r * LDA + c4 + 0] = (__bf16)pa[t].x;
            As[r * LDA + c4 + 1] = (__bf16)pa[t].y;
            As[r * LDA + c4 + 2] = (__bf16)pa[t].z;
            As[r * LDA + c4 + 3] = (__bf16)pa[t].w;
        }
        {
            int k = tid & 31, ng = tid >> 5;
            float vv[16] = {pw[0].x, pw[0].y, pw[0].z, pw[0].w,
                            pw[1].x, pw[1].y, pw[1].z, pw[1].w,
                            pw[2].x, pw[2].y, pw[2].z, pw[2].w,
                            pw[3].x, pw[3].y, pw[3].z, pw[3].w};
#pragma unroll
            for (int i = 0; i < 16; ++i)
                Bs[(ng * 16 + i) * LDB + k] = (__bf16)vv[i];
        }
        __syncthreads();
        if (kb + BK < FEAT) load_tile(kb + BK);   // overlap next tile with WMMA

        v16bf a[2], b[4];
        const int bk0 = (lane >> 4) << 4;
#pragma unroll
        for (int tm = 0; tm < 2; ++tm)
            a[tm] = load_a_frag(&As[(wr * 32 + tm * 16 + (lane & 15)) * LDA], lane);
#pragma unroll
        for (int tn = 0; tn < 4; ++tn)
            b[tn] = load_b_frag(&Bs[(wc * 64 + tn * 16 + (lane & 15)) * LDB + bk0]);
#pragma unroll
        for (int tm = 0; tm < 2; ++tm)
#pragma unroll
            for (int tn = 0; tn < 4; ++tn)
                acc[tm][tn] = __builtin_amdgcn_wmma_f32_16x16x32_bf16(
                    false, a[tm], false, b[tn], (short)0, acc[tm][tn], false, false);
    }

    // ---- epilogue: fold column scale d[j], store transposed bf16 ----
    const int nn = lane & 15;
    const int mb = (lane >> 4) << 3;
#pragma unroll
    for (int tm = 0; tm < 2; ++tm) {
#pragma unroll
        for (int r = 0; r < 8; ++r) {
            int gRow = rb + wr * 32 + tm * 16 + mb + r;
            float dv = dinv[gRow];
#pragma unroll
            for (int tn = 0; tn < 4; ++tn) {
                int gCol = cb + wc * 64 + tn * 16 + nn;
                supportT[(size_t)gCol * N_NODES + gRow] = (__bf16)(dv * acc[tm][tn][r]);
            }
        }
    }
}

// ---------------------------------------------------------------------------
// Kernel 3: out[i,f] = relu( d[i] * sum_j (adj[i,j]+I) * d[j]*support[j,f] )
// A: fp32->bf16 (+diag) through VGPRs; B: async global->LDS raw bf16 copy
// into a double-buffered LDS tile (ASYNCcnt), overlapped with WMMA.
// ---------------------------------------------------------------------------
__global__ __launch_bounds__(256)
void gcn_agg_kernel(const float* __restrict__ adj,
                    const __bf16* __restrict__ supportT,
                    const float* __restrict__ dinv,
                    float* __restrict__ out) {
    __shared__ __bf16 As[BM * LDA];        // adj+I tile, bf16
    __shared__ __bf16 Bs[2][BN * LDB];     // supportT tile [n][k], double-buffered

    const int tid  = threadIdx.x;
    const int wave = tid >> 5, lane = tid & 31;
    const int wr = wave & 3, wc = wave >> 2;
    const int cb = blockIdx.x * BN;            // x = column tile (L2 band reuse)
    const int rb = blockIdx.y * BM;

    const int bn  = tid >> 1;                  // 0..127: column within B tile
    const int bkh = (tid & 1) << 4;            // 0 or 16: K half

    float4 pa[4];
    auto load_a_tile = [&](int kb) {
#pragma unroll
        for (int t = 0; t < 4; ++t) {
            int idx = tid + t * 256;
            int r = idx >> 3, c4 = (idx & 7) << 2;
            pa[t] = *reinterpret_cast<const float4*>(
                adj + (size_t)(rb + r) * N_NODES + kb + c4);
        }
    };
    auto fill_b_tile = [&](int kb, int buf) {
        const __bf16* src = supportT + (size_t)(cb + bn) * N_NODES + kb + bkh;
        __bf16* dst = &Bs[buf][bn * LDB + bkh];
#if USE_ASYNC_LDS
        // global_load_async_to_lds_b128: IOFFSET applies to both addresses
        __builtin_amdgcn_global_load_async_to_lds_b128((v4i*)src, (v4i*)dst, 0, 0);
        __builtin_amdgcn_global_load_async_to_lds_b128((v4i*)src, (v4i*)dst, 16, 0);
#else
        const uint4* s4 = reinterpret_cast<const uint4*>(src);
        uint4* d4 = reinterpret_cast<uint4*>(dst);
        d4[0] = s4[0];
        d4[1] = s4[1];
#endif
    };

    v8f acc[2][4] = {};
    load_a_tile(0);
    fill_b_tile(0, 0);

    for (int kb = 0; kb < N_NODES; kb += BK) {
        const int p = (kb >> 5) & 1;           // current B buffer
#if USE_ASYNC_LDS
        __builtin_amdgcn_s_wait_asynccnt(0);   // our async fills of Bs[p] done
#endif
        __syncthreads();                       // all waves' fills visible; As free
        // ---- A tile: fp32 -> bf16, +1 on diagonal ----
#pragma unroll
        for (int t = 0; t < 4; ++t) {
            int idx = tid + t * 256;
            int r = idx >> 3, c4 = (idx & 7) << 2;
            float f[4] = {pa[t].x, pa[t].y, pa[t].z, pa[t].w};
#pragma unroll
            for (int i = 0; i < 4; ++i) {
                float fv = f[i] + ((rb + r) == (kb + c4 + i) ? 1.0f : 0.0f);
                As[r * LDA + c4 + i] = (__bf16)fv;
            }
        }
        if (kb + BK < N_NODES) {
            load_a_tile(kb + BK);              // global loads overlap WMMA below
            fill_b_tile(kb + BK, 1 - p);       // async fill overlaps WMMA below
        }
        __syncthreads();

        v16bf a[2], b[4];
        const int bk0 = (lane >> 4) << 4;
#pragma unroll
        for (int tm = 0; tm < 2; ++tm)
            a[tm] = load_a_frag(&As[(wr * 32 + tm * 16 + (lane & 15)) * LDA], lane);
#pragma unroll
        for (int tn = 0; tn < 4; ++tn)
            b[tn] = load_b_frag(&Bs[p][(wc * 64 + tn * 16 + (lane & 15)) * LDB + bk0]);
#pragma unroll
        for (int tm = 0; tm < 2; ++tm)
#pragma unroll
            for (int tn = 0; tn < 4; ++tn)
                acc[tm][tn] = __builtin_amdgcn_wmma_f32_16x16x32_bf16(
                    false, a[tm], false, b[tn], (short)0, acc[tm][tn], false, false);
    }

    // ---- epilogue: row scale d[i] then ReLU, fp32 out ----
    const int nn = lane & 15;
    const int mb = (lane >> 4) << 3;
#pragma unroll
    for (int tm = 0; tm < 2; ++tm) {
#pragma unroll
        for (int r = 0; r < 8; ++r) {
            int gRow = rb + wr * 32 + tm * 16 + mb + r;
            float dv = dinv[gRow];
#pragma unroll
            for (int tn = 0; tn < 4; ++tn) {
                int gCol = cb + wc * 64 + tn * 16 + nn;
                float v = dv * acc[tm][tn][r];
                out[(size_t)gRow * FEAT + gCol] = v > 0.f ? v : 0.f;
            }
        }
    }
}

// ---------------------------------------------------------------------------
// Launch: degree -> (x@W)*d (transposed bf16) -> normalized aggregate + ReLU
// Workspace: [0,32KB) fp32 dinv[8192]; then bf16 supportT[512*8192] (8MB)
// ---------------------------------------------------------------------------
extern "C" void kernel_launch(void* const* d_in, const int* in_sizes, int n_in,
                              void* d_out, int out_size, void* d_ws, size_t ws_size,
                              hipStream_t stream) {
    (void)in_sizes; (void)n_in; (void)out_size; (void)ws_size;

    const float* x   = (const float*)d_in[0];   // [8192, 512]
    const float* adj = (const float*)d_in[1];   // [8192, 8192]
    const float* w   = (const float*)d_in[2];   // [512, 512]
    float* out = (float*)d_out;                 // [8192, 512]

    float*  dinv     = (float*)d_ws;
    __bf16* supportT = (__bf16*)((char*)d_ws + (size_t)N_NODES * sizeof(float));

    gcn_degree_kernel<<<N_NODES, 256, 0, stream>>>(adj, dinv);

    dim3 grid(FEAT / BN, N_NODES / BM);   // (4, 64): x = col tile -> L2 band reuse
    gcn_xw_kernel<<<grid, 256, 0, stream>>>(x, w, dinv, supportT);
    gcn_agg_kernel<<<grid, 256, 0, stream>>>(adj, supportT, dinv, out);
}
